// DSSTGNN_49435073577512
// MI455X (gfx1250) — compile-verified, matrix operand order checked by reading, not run on hardware
//
#include <hip/hip_runtime.h>
#include <hip/hip_bf16.h>

typedef _Float16 f16;
typedef __attribute__((ext_vector_type(16))) _Float16 v16h;
typedef __attribute__((ext_vector_type(8)))  _Float16 v8h;
typedef __attribute__((ext_vector_type(8)))  float    v8f;

#define B_      512
#define CIN_    5
#define N_      156
#define T_      12
#define CH_     64
#define NC_     128
#define TIME_   288
#define M_TOT   (B_*N_*T_)        // 958464, divisible by 16 and 256
#define PRED_SZ (B_*3*N_*12)      // 2875392
#define GG_SZ   (N_*N_)           // 24336
#define TPW     8                 // M-tiles per wave in the DSTE kernel

__device__ inline v8f zero8f() {
  v8f v;
  #pragma unroll
  for (int i = 0; i < 8; ++i) v[i] = 0.f;
  return v;
}
__device__ inline v8h zero8h() {
  v8h v;
  #pragma unroll
  for (int i = 0; i < 8; ++i) v[i] = (f16)0;
  return v;
}

// ---------------------------------------------------------------------------
// K0: Haar level-2 DWT split + two K=5 start convs -> H0[m][128] (f16)
//     channels 0..63 = s1(xl), 64..127 = s2(xh);  m = (b*N+n)*T + t
// ---------------------------------------------------------------------------
__global__ __launch_bounds__(256) void k_dwt_start(
    const float* __restrict__ x,
    const float* __restrict__ w1, const float* __restrict__ b1,
    const float* __restrict__ w2, const float* __restrict__ b2,
    f16* __restrict__ H0)
{
  __shared__ float sw1[CH_*CIN_], sb1[CH_], sw2[CH_*CIN_], sb2[CH_];
  const int tid = threadIdx.x;
  for (int i = tid; i < CH_*CIN_; i += 256) { sw1[i] = w1[i]; sw2[i] = w2[i]; }
  if (tid < CH_) { sb1[tid] = b1[tid]; sb2[tid] = b2[tid]; }
  __syncthreads();

  const int m  = blockIdx.x*256 + tid;
  const int t  = m % T_;
  const int bn = m / T_;
  const int n  = bn % N_;
  const int b  = bn / N_;
  const int tb = t >> 2;

  float xl[CIN_], xh[CIN_];
  #pragma unroll
  for (int c = 0; c < CIN_; ++c) {
    const float* xp = x + (((size_t)(b*CIN_ + c)*N_ + n)*T_);
    float s = 0.f;
    #pragma unroll
    for (int j = 0; j < 4; ++j) s += xp[tb*4 + j];
    s *= 0.25f;
    xl[c] = s;
    xh[c] = xp[t] - s;
  }
  f16* out = H0 + (size_t)m*128;
  for (int ch = 0; ch < CH_; ++ch) {
    float a1 = sb1[ch], a2 = sb2[ch];
    #pragma unroll
    for (int c = 0; c < CIN_; ++c) {
      a1 += sw1[ch*CIN_+c]*xl[c];
      a2 += sw2[ch*CIN_+c]*xh[c];
    }
    out[ch]     = (f16)a1;
    out[64+ch]  = (f16)a2;
  }
}

// ---------------------------------------------------------------------------
// K1: gated dilated temporal conv layer as a 256x256xM WMMA GEMM.
//     Y[o,m] = sum_k<128 Wcat[o,k]*Hin[m-d,k] + sum_k>=128 Wcat[o,k]*Hin[m,k-128] + b[o]
//     Hout[m,c] = tanh(Y[c,m]) * sigmoid(Y[c+128,m])
//     Each wave processes TPW consecutive 16-wide M tiles to amortize the
//     one-time LDS weight staging; a- and g- output tiles computed together.
// ---------------------------------------------------------------------------
__global__ __launch_bounds__(256) void k_dste_layer(
    const f16* __restrict__ Hin, f16* __restrict__ Hout,
    const float* __restrict__ w, const float* __restrict__ bias, int dil)
{
  __shared__ __attribute__((aligned(16))) f16 Wl[256*256];   // 128 KB
  __shared__ float Bl[256];
  const int tid = threadIdx.x;

  // Stage Wcat (f32 -> f16). dste_w shape (256,128,2): tap0 -> shifted, tap1 -> h.
  for (int i = tid; i < 256*256; i += 256) {
    const int o = i >> 8, k = i & 255;
    const float v = (k < 128) ? w[(o*128 + k)*2] : w[(o*128 + (k-128))*2 + 1];
    Wl[i] = (f16)v;
  }
  Bl[tid] = bias[tid];
  __syncthreads();

  const int wave  = tid >> 5;
  const int lane  = tid & 31;
  const int g     = lane >> 4;         // lane group (K sub-pattern / row half)
  const int tile0 = (blockIdx.x*8 + wave) * TPW;

  #pragma unroll 1
  for (int rep = 0; rep < TPW; ++rep) {
    const int tile = tile0 + rep;
    const int mcol = (tile << 4) + (lane & 15);
    const int t    = mcol % T_;
    const bool havePrev = (t >= dil);

    const f16* rowH  = Hin + (size_t)mcol*128;
    const f16* rowHp = havePrev ? (Hin + (size_t)(mcol - dil)*128) : rowH;

    // Hint the next tile's activation row into cache (global_prefetch_b8).
    if (rep + 1 < TPW) __builtin_prefetch(rowH + 16*128, 0, 0);

    // Preload the 8 B fragments (K = 256 in 8 steps of 32). Per-lane K pattern
    // for 16-bit fragments: runs [8g,8g+8) and [16+8g,16+8g+8) -> two 16B loads.
    v16h bf[8];
    #pragma unroll
    for (int kk = 0; kk < 8; ++kk) {
      const f16* src; int c0; bool ok;
      if (kk < 4) { src = rowHp; c0 = kk*32 + 8*g;       ok = havePrev; }
      else        { src = rowH;  c0 = (kk-4)*32 + 8*g;   ok = true;     }
      v8h lo = zero8h(), hi = zero8h();
      if (ok) {
        lo = *reinterpret_cast<const v8h*>(src + c0);
        hi = *reinterpret_cast<const v8h*>(src + c0 + 16);
      }
      bf[kk] = __builtin_shufflevector(lo, hi, 0,1,2,3,4,5,6,7,8,9,10,11,12,13,14,15);
    }

    #pragma unroll 1
    for (int ot = 0; ot < 8; ++ot) {
      const int oa = ot*16 + (lane & 15);   // 'a' output row (channel 0..127)
      const int og = oa + 128;              // paired 'g' output row
      v8f acc_a = zero8f();
      v8f acc_g = zero8f();
      #pragma unroll
      for (int kk = 0; kk < 8; ++kk) {
        const f16* wa = Wl + (size_t)oa*256 + kk*32 + 8*g;
        const f16* wg = Wl + (size_t)og*256 + kk*32 + 8*g;
        v8h la = *reinterpret_cast<const v8h*>(wa);
        v8h ha = *reinterpret_cast<const v8h*>(wa + 16);
        v8h lg = *reinterpret_cast<const v8h*>(wg);
        v8h hg = *reinterpret_cast<const v8h*>(wg + 16);
        v16h fa = __builtin_shufflevector(la, ha, 0,1,2,3,4,5,6,7,8,9,10,11,12,13,14,15);
        v16h fg = __builtin_shufflevector(lg, hg, 0,1,2,3,4,5,6,7,8,9,10,11,12,13,14,15);
        acc_a = __builtin_amdgcn_wmma_f32_16x16x32_f16(false, fa, false, bf[kk],
                                                       (short)0, acc_a, false, false);
        acc_g = __builtin_amdgcn_wmma_f32_16x16x32_f16(false, fg, false, bf[kk],
                                                       (short)0, acc_g, false, false);
      }
      // D layout: VGPR r holds row (channel) ot*16 + 8*g + r, column mcol.
      const int chbase = ot*16 + 8*g;
      v8h out8;
      #pragma unroll
      for (int r = 0; r < 8; ++r) {
        const float a  = acc_a[r] + Bl[chbase + r];
        const float gg = acc_g[r] + Bl[chbase + 128 + r];
        const float hv = tanhf(a) * (1.f / (1.f + __expf(-gg)));
        out8[r] = (f16)hv;
      }
      *reinterpret_cast<v8h*>(Hout + (size_t)mcol*128 + chbase) = out8;
    }
  }
}

// ---------------------------------------------------------------------------
// K2: feat = Wout @ h2[.., T-1] ; TE from embedding gathers; data_st=[feat;TE];
//     skip = fc_st @ data_st + b. One block per (b,n), 128 threads.
// ---------------------------------------------------------------------------
__global__ __launch_bounds__(128) void k_feat_te(
    const float* __restrict__ x, const f16* __restrict__ H2,
    const float* __restrict__ wout, const float* __restrict__ bout,
    const float* __restrict__ tday, const float* __restrict__ tweek,
    const float* __restrict__ fcw,  const float* __restrict__ fcb,
    float* __restrict__ data_st, float* __restrict__ skip)
{
  __shared__ float h2[128];
  __shared__ float ds[128];
  const int idx = blockIdx.x;          // b*N + n
  const int n = idx % N_, b = idx / N_;
  const int c = threadIdx.x;
  h2[c] = (float)H2[((size_t)idx*T_ + (T_-1))*128 + c];
  __syncthreads();

  float v;
  if (c < 64) {
    float a = bout[c];
    const float* wr = wout + c*128;
    for (int k = 0; k < 128; ++k) a += wr[k]*h2[k];
    v = a;
  } else {
    const float td = x[(((size_t)b*CIN_ + 1)*N_ + n)*T_ + (T_-1)];
    const float dw = x[(((size_t)b*CIN_ + 2)*N_ + n)*T_ + (T_-1)];
    int tod = (int)(td * TIME_); tod = min(max(tod, 0), TIME_-1);
    int dow = (int)dw;           dow = min(max(dow, 0), 6);
    v = tday[tod*64 + (c-64)] + tweek[dow*64 + (c-64)];
  }
  ds[c] = v;
  data_st[(size_t)idx*128 + c] = v;
  __syncthreads();

  float sk = fcb[c];
  const float* fr = fcw + c*128;
  for (int k = 0; k < 128; ++k) sk += fr[k]*ds[k];
  skip[(size_t)idx*128 + c] = sk;
}

// ---------------------------------------------------------------------------
// K3: E_d[6,N] from batch element B-1 only (reference indexes [-1]).
// ---------------------------------------------------------------------------
__global__ __launch_bounds__(192) void k_ed(
    const float* __restrict__ x, const float* __restrict__ tday,
    const float* __restrict__ tweek,
    const float* __restrict__ mw,  const float* __restrict__ mb,
    const float* __restrict__ fdw, const float* __restrict__ fdb,
    const float* __restrict__ fww, const float* __restrict__ fwb,
    const float* __restrict__ Es,  float* __restrict__ Ed)
{
  const int n = threadIdx.x;
  if (n >= N_) return;
  const int b = B_-1;
  float xc[CIN_];
  #pragma unroll
  for (int c = 0; c < CIN_; ++c)
    xc[c] = x[(((size_t)b*CIN_ + c)*N_ + n)*T_ + (T_-1)];
  int tod = (int)(xc[1]*TIME_); tod = min(max(tod, 0), TIME_-1);
  int dow = (int)xc[2];         dow = min(max(dow, 0), 6);
  const float* ETd = tday  + tod*64;
  const float* EDw = tweek + dow*64;
  for (int d = 0; d < 6; ++d) {
    float m = mb[d];
    #pragma unroll
    for (int c = 0; c < CIN_; ++c) m += mw[d*CIN_+c]*xc[c];
    float a = fdb[d], w = fwb[d];
    for (int k = 0; k < 64; ++k) { a += fdw[d*64+k]*ETd[k]; w += fww[d*64+k]*EDw[k]; }
    Ed[d*N_+n] = tanhf(m * a * w * Es[d*N_+n]);
  }
}

// ---------------------------------------------------------------------------
// K4: D_graph = softmax(relu(Ed^T Ed), axis=1); A_graph = softmax(relu(nv1 nv2^T),1)
//     S = D + A. One block per row n; 256 threads (>=156) for the reductions.
// ---------------------------------------------------------------------------
__global__ __launch_bounds__(256) void k_graphs(
    const float* __restrict__ Ed, const float* __restrict__ nv1,
    const float* __restrict__ nv2,
    float* __restrict__ Aout, float* __restrict__ Dout, float* __restrict__ S)
{
  __shared__ float rA[256], rD[256];
  const int n = blockIdx.x, m = threadIdx.x;
  float gd = 0.f, ga = 0.f;
  if (m < N_) {
    #pragma unroll
    for (int d = 0; d < 6; ++d) gd += Ed[d*N_+n]*Ed[d*N_+m];
    #pragma unroll
    for (int k = 0; k < 10; ++k) ga += nv1[n*10+k]*nv2[m*10+k];
    gd = fmaxf(gd, 0.f); ga = fmaxf(ga, 0.f);
  }
  rA[m] = (m < N_) ? ga : -1e30f;
  rD[m] = (m < N_) ? gd : -1e30f;
  __syncthreads();
  for (int s = 128; s > 0; s >>= 1) {
    if (m < s) { rA[m] = fmaxf(rA[m], rA[m+s]); rD[m] = fmaxf(rD[m], rD[m+s]); }
    __syncthreads();
  }
  const float mA = rA[0], mD = rD[0];
  __syncthreads();
  const float eA = (m < N_) ? __expf(ga - mA) : 0.f;
  const float eD = (m < N_) ? __expf(gd - mD) : 0.f;
  rA[m] = eA; rD[m] = eD;
  __syncthreads();
  for (int s = 128; s > 0; s >>= 1) {
    if (m < s) { rA[m] += rA[m+s]; rD[m] += rD[m+s]; }
    __syncthreads();
  }
  const float sA = rA[0], sD = rD[0];
  if (m < N_) {
    const float av = eA / sA, dv = eD / sD;
    Aout[n*N_+m] = av;
    Dout[n*N_+m] = dv;
    S[n*N_+m]    = av + dv;
  }
}

// ---------------------------------------------------------------------------
// K5: graph conv layer: agg = (D+A) @ xg (over nodes) ; xg = relu(W agg + b) + xg
//     One block per batch element: stage xg slice + weights in LDS.
// ---------------------------------------------------------------------------
__global__ __launch_bounds__(128) void k_gconv(
    const float* __restrict__ Xin, float* __restrict__ Xout,
    const float* __restrict__ S, const float* __restrict__ w,
    const float* __restrict__ bias)
{
  __shared__ float xg[N_*128];      // 78 KB
  __shared__ float wl[128*128];     // 64 KB
  __shared__ float bl[128];
  __shared__ float srow[N_];
  __shared__ float agg[128];
  const int b = blockIdx.x, tid = threadIdx.x;
  const float* Xb = Xin + (size_t)b*N_*128;
  for (int i = tid; i < N_*128; i += 128)  xg[i] = Xb[i];
  for (int i = tid; i < 128*128; i += 128) wl[i] = w[i];
  bl[tid] = bias[tid];
  __syncthreads();

  for (int n = 0; n < N_; ++n) {
    for (int i = tid; i < N_; i += 128) srow[i] = S[n*N_+i];
    __syncthreads();
    float a = 0.f;
    for (int mm = 0; mm < N_; ++mm) a += srow[mm]*xg[mm*128+tid];
    agg[tid] = a;
    __syncthreads();
    float o = bl[tid];
    const float* wr = wl + tid*128;
    for (int k = 0; k < 128; ++k) o += wr[k]*agg[k];
    o = fmaxf(o, 0.f) + xg[n*128+tid];
    Xout[((size_t)b*N_ + n)*128 + tid] = o;
    __syncthreads();
  }
}

// ---------------------------------------------------------------------------
// K6: pred = reg_w @ (xg + skip) + reg_b, with the (12,3) channel de-interleave
//     into output layout [B,3,N,12].
// ---------------------------------------------------------------------------
__global__ __launch_bounds__(128) void k_pred(
    const float* __restrict__ Xg, const float* __restrict__ Skip,
    const float* __restrict__ rw, const float* __restrict__ rb,
    float* __restrict__ out)
{
  __shared__ float dsf[128];
  const int idx = blockIdx.x;
  const int n = idx % N_, b = idx / N_;
  const int c = threadIdx.x;
  dsf[c] = Xg[(size_t)idx*128 + c] + Skip[(size_t)idx*128 + c];
  __syncthreads();
  if (c < 36) {
    float v = rb[c];
    const float* wr = rw + c*128;
    for (int k = 0; k < 128; ++k) v += wr[k]*dsf[k];
    const int t = c / 3, r = c % 3;
    out[(((size_t)b*3 + r)*N_ + n)*12 + t] = v;
  }
}

// ---------------------------------------------------------------------------
extern "C" void kernel_launch(void* const* d_in, const int* in_sizes, int n_in,
                              void* d_out, int out_size, void* d_ws, size_t ws_size,
                              hipStream_t stream) {
  const float* x        = (const float*)d_in[0];
  const float* w_start1 = (const float*)d_in[1];
  const float* b_start1 = (const float*)d_in[2];
  const float* w_start2 = (const float*)d_in[3];
  const float* b_start2 = (const float*)d_in[4];
  const float* dste_w0  = (const float*)d_in[5];
  const float* dste_b0  = (const float*)d_in[6];
  const float* dste_w1  = (const float*)d_in[7];
  const float* dste_b1  = (const float*)d_in[8];
  const float* dste_wout= (const float*)d_in[9];
  const float* dste_bout= (const float*)d_in[10];
  const float* time_day = (const float*)d_in[11];
  const float* time_week= (const float*)d_in[12];
  const float* mlp_w    = (const float*)d_in[13];
  const float* mlp_b    = (const float*)d_in[14];
  const float* fcd_w    = (const float*)d_in[15];
  const float* fcd_b    = (const float*)d_in[16];
  const float* fcw_w    = (const float*)d_in[17];
  const float* fcw_b    = (const float*)d_in[18];
  const float* E_s      = (const float*)d_in[19];
  const float* fc_st_w  = (const float*)d_in[20];
  const float* fc_st_b  = (const float*)d_in[21];
  const float* nodevec1 = (const float*)d_in[22];
  const float* nodevec2 = (const float*)d_in[23];
  const float* gconv_w0 = (const float*)d_in[24];
  const float* gconv_b0 = (const float*)d_in[25];
  const float* gconv_w1 = (const float*)d_in[26];
  const float* gconv_b1 = (const float*)d_in[27];
  const float* reg_w    = (const float*)d_in[28];
  const float* reg_b    = (const float*)d_in[29];

  char* ws = (char*)d_ws;
  const size_t SZH = (size_t)M_TOT * 128 * sizeof(f16);      // 245,366,784
  const size_t SD  = (size_t)B_ * N_ * 128 * sizeof(float);  //  40,894,464
  f16*   H0      = (f16*)(ws);
  f16*   H1      = (f16*)(ws + SZH);
  // Post-DSTE tensors reuse the (dead) H1 region.
  float* data_st = (float*)(ws + SZH);
  float* skip    = (float*)(ws + SZH + SD);
  float* xgA     = (float*)(ws + SZH + 2*SD);
  float* xgB     = (float*)(ws + SZH + 3*SD);
  float* Ed      = (float*)(ws + 2*SZH);
  float* Ssum    = (float*)(ws + 2*SZH + 4096);

  float* out   = (float*)d_out;
  float* Aout  = out + PRED_SZ;
  float* Dout  = out + PRED_SZ + GG_SZ;

  k_dwt_start<<<M_TOT/256, 256, 0, stream>>>(x, w_start1, b_start1,
                                             w_start2, b_start2, H0);
  k_dste_layer<<<(M_TOT/16)/(8*TPW), 256, 0, stream>>>(H0, H1, dste_w0, dste_b0, 1);
  k_dste_layer<<<(M_TOT/16)/(8*TPW), 256, 0, stream>>>(H1, H0, dste_w1, dste_b1, 2);
  k_feat_te<<<B_*N_, 128, 0, stream>>>(x, H0, dste_wout, dste_bout,
                                       time_day, time_week, fc_st_w, fc_st_b,
                                       data_st, skip);
  k_ed<<<1, 192, 0, stream>>>(x, time_day, time_week, mlp_w, mlp_b,
                              fcd_w, fcd_b, fcw_w, fcw_b, E_s, Ed);
  k_graphs<<<N_, 256, 0, stream>>>(Ed, nodevec1, nodevec2, Aout, Dout, Ssum);
  k_gconv<<<B_, 128, 0, stream>>>(data_st, xgA, Ssum, gconv_w0, gconv_b0);
  k_gconv<<<B_, 128, 0, stream>>>(xgA, xgB, Ssum, gconv_w1, gconv_b1);
  k_pred<<<B_*N_, 128, 0, stream>>>(xgB, skip, reg_w, reg_b, out);
}
